// MultiHeadAttention_31576599561118
// MI455X (gfx1250) — matrix operand order repaired and checked by reading, not compiled, not run on hardware
//
#include <hip/hip_runtime.h>
#include <math.h>

// ---------------------------------------------------------------------------
// MultiHeadAttention for MI455X (gfx1250): f16 WMMA everywhere, f32 accumulate.
//   1) prep: x -> f16, w_qkv/w_proj -> transposed f16 [K][N]
//   2) QKV GEMM (wave = 32x64 tile, B frags reused x2) -> Q/Kt/V head layouts
//   3) flash attention (wave = 16 queries, 32-key blocks, online softmax)
//   4) output projection GEMM (wave = 32x64) + bias -> f32 d_out
// Workspace use: 48 MiB.
// ---------------------------------------------------------------------------

typedef _Float16 h16;
typedef __attribute__((ext_vector_type(16))) _Float16 v16h;
typedef __attribute__((ext_vector_type(8)))  _Float16 v8h;
typedef __attribute__((ext_vector_type(8)))  float    v8f;

#define D_IN    1024
#define D_OUT   1024
#define NHEADS  16
#define HDIM    64
#define SEQ     2048
#define BATCH   2
#define MROWS   (BATCH * SEQ)   // 4096
#define NQKV    (3 * D_OUT)     // 3072

static __device__ __forceinline__ v8f wmma16x16x32(v16h a, v16h b, v8f c) {
    // D = A(16x32) * B(32x16) + C, f32 accumulate
    return __builtin_amdgcn_wmma_f32_16x16x32_f16(false, a, false, b,
                                                  (short)0, c, false, false);
}

// A-fragment (16x32, M x K), row-major source with leading dim ld.
// lane<16:  row=lane,    K = k0+0..7   and k0+16..23
// lane>=16: row=lane-16, K = k0+8..15  and k0+24..31
static __device__ __forceinline__ v16h load_a_frag(const h16* __restrict__ base,
                                                   int ld, int row0, int k0, int lane) {
    const int half = lane >> 4, lr = lane & 15;
    const h16* p = base + (size_t)(row0 + lr) * ld + k0 + half * 8;
    v8h lo = *(const v8h*)p;
    v8h hi = *(const v8h*)(p + 16);
    v16h a;
#pragma unroll
    for (int i = 0; i < 8; ++i) { a[i] = lo[i]; a[8 + i] = hi[i]; }
    return a;
}

// B-fragment (32x16, K x N), row-major K x N source with leading dim ld.
// lane l holds column N = n0+(l&15); K = k0+8h+0..7 and k0+16+8h+0..7 (h = l>>4).
static __device__ __forceinline__ v16h load_b_frag(const h16* __restrict__ base,
                                                   int ld, int k0, int n0, int lane) {
    const int half = lane >> 4, lr = lane & 15;
    const h16* p = base + (size_t)(k0 + half * 8) * ld + n0 + lr;
    v16h b;
#pragma unroll
    for (int i = 0; i < 8; ++i) { b[i] = p[(size_t)i * ld]; b[8 + i] = p[(size_t)(16 + i) * ld]; }
    return b;
}

// ---------------------------- prep kernels ---------------------------------

__global__ void convert_f32_to_f16_kernel(const float* __restrict__ in,
                                          h16* __restrict__ out, int n) {
    int i = blockIdx.x * blockDim.x + threadIdx.x;
    if (i < n) out[i] = (h16)in[i];
}

// in: [ROWS][COLS] f32 -> out: [COLS][ROWS] f16 (COLS is a power of two)
template <int ROWS, int COLS>
__global__ void transpose_to_f16_kernel(const float* __restrict__ in,
                                        h16* __restrict__ out) {
    int i = blockIdx.x * blockDim.x + threadIdx.x;
    if (i < ROWS * COLS) {
        int r = i >> __builtin_ctz(COLS);
        int c = i & (COLS - 1);                         // coalesced read
        out[(size_t)c * ROWS + r] = (h16)in[i];
    }
}

// ---------------------------- QKV projection -------------------------------
// xb [4096][1024] f16, wT [1024][3072] f16 (already transposed).
// Each wave computes a 32(M) x 64(N) tile: 2 A fragments, 4 B fragments per
// K-step, each B fragment feeds 2 WMMAs (register reuse -> 2x arithmetic
// intensity vs 16x64 blocking). Block = 4 waves = 128x64 tile.
// Output scattered into head layouts: Q[b,h,s,d], Kt[b,h,d,s], V[b,h,s,d].

__global__ __launch_bounds__(128) void qkv_gemm_kernel(
        const h16* __restrict__ xb, const h16* __restrict__ wT,
        h16* __restrict__ Qb, h16* __restrict__ Ktb, h16* __restrict__ Vb) {
    const int wave = threadIdx.x >> 5;
    const int lane = threadIdx.x & 31;
    const int n0 = blockIdx.x * 64;
    const int m0 = blockIdx.y * 128 + wave * 32;

    v8f acc[2][4] = {};
    for (int k = 0; k < D_IN; k += 32) {
        v16h a0 = load_a_frag(xb, D_IN, m0,      k, lane);
        v16h a1 = load_a_frag(xb, D_IN, m0 + 16, k, lane);
#pragma unroll
        for (int j = 0; j < 4; ++j) {
            v16h b = load_b_frag(wT, NQKV, k, n0 + 16 * j, lane);
            acc[0][j] = wmma16x16x32(a0, b, acc[0][j]);
            acc[1][j] = wmma16x16x32(a1, b, acc[1][j]);
        }
    }

    const int half = lane >> 4, lr = lane & 15;
#pragma unroll
    for (int i = 0; i < 2; ++i) {
        const int mb = m0 + i * 16;
#pragma unroll
        for (int j = 0; j < 4; ++j) {
            const int n = n0 + 16 * j + lr;        // column in [0,3072)
            const int t = n >> 10;                 // 0=Q 1=K 2=V
            const int rem = n & 1023;
            const int h = rem >> 6;                // head
            const int d = rem & 63;                // dim in head
#pragma unroll
            for (int r = 0; r < 8; ++r) {
                const int m = mb + r + 8 * half;   // row in [0,4096)
                const int bb = m >> 11;            // batch
                const int s  = m & (SEQ - 1);
                const h16 val = (h16)acc[i][j][r];
                const size_t bh = (size_t)(bb * NHEADS + h);
                if (t == 0)      Qb [(bh * SEQ + s) * HDIM + d] = val;
                else if (t == 1) Ktb[(bh * HDIM + d) * SEQ + s] = val;
                else             Vb [(bh * SEQ + s) * HDIM + d] = val;
            }
        }
    }
}

// ---------------------------- flash attention ------------------------------
// One wave per (b, h, 16-query tile). Keys processed in blocks of 32 with
// causal masking + online softmax. P tile goes through LDS to convert the
// C-fragment layout into an A-fragment layout for the P@V WMMA.

__global__ __launch_bounds__(128) void attn_kernel(
        const h16* __restrict__ Qb, const h16* __restrict__ Ktb,
        const h16* __restrict__ Vb, h16* __restrict__ ctxb) {
    __shared__ __align__(16) h16 plds[4][16 * 32];   // 1 KB per wave

    const int wave = threadIdx.x >> 5;
    const int lane = threadIdx.x & 31;
    const int half = lane >> 4, lr = lane & 15;

    const int W  = blockIdx.x * 4 + wave;   // 0 .. B*H*(S/16)-1
    const int qt = W & 127;                 // S/16 = 128 query tiles
    const int bh = W >> 7;                  // 0 .. 31
    const int b  = bh >> 4, h = bh & 15;
    const int q0 = qt * 16;

    const h16* Qp  = Qb  + (size_t)bh * SEQ * HDIM;
    const h16* Ktp = Ktb + (size_t)bh * HDIM * SEQ;
    const h16* Vp  = Vb  + (size_t)bh * SEQ * HDIM;
    h16* pl = plds[wave];

    // Q fragments for both K-steps of the 64-wide head dim (kept in VGPRs)
    const v16h qa0 = load_a_frag(Qp, HDIM, q0, 0,  lane);
    const v16h qa1 = load_a_frag(Qp, HDIM, q0, 32, lane);

    v8f acc[4] = {};
    float mrow[8], lrow[8];
#pragma unroll
    for (int r = 0; r < 8; ++r) { mrow[r] = -3.0e38f; lrow[r] = 0.0f; }
    const float scale = 0.125f;             // 1/sqrt(64)
    const int qmax = q0 + 15;

    for (int j0 = 0; j0 <= qmax; j0 += 32) {
        // ---- scores: 16 queries x 32 keys, as two 16x16 C tiles ----
        v8f s[2];
#pragma unroll
        for (int sh = 0; sh < 2; ++sh) {
            v16h kb0 = load_b_frag(Ktp, SEQ, 0,  j0 + sh * 16, lane);
            v16h kb1 = load_b_frag(Ktp, SEQ, 32, j0 + sh * 16, lane);
            v8f sc = {};
            sc = wmma16x16x32(qa0, kb0, sc);
            sc = wmma16x16x32(qa1, kb1, sc);
            s[sh] = sc;
        }

        // ---- scale, causal mask, online softmax update ----
#pragma unroll
        for (int r = 0; r < 8; ++r) {
            const int q = q0 + r + 8 * half;
            float v0 = s[0][r] * scale; if (j0 + lr      > q) v0 = -3.0e38f;
            float v1 = s[1][r] * scale; if (j0 + 16 + lr > q) v1 = -3.0e38f;

            float vmax = fmaxf(v0, v1);
#pragma unroll
            for (int off = 8; off >= 1; off >>= 1)
                vmax = fmaxf(vmax, __shfl_xor(vmax, off, 32));
            const float mnew  = fmaxf(mrow[r], vmax);
            const float alpha = __expf(mrow[r] - mnew);
            const float p0 = __expf(v0 - mnew);
            const float p1 = __expf(v1 - mnew);

            float ps = p0 + p1;
#pragma unroll
            for (int off = 8; off >= 1; off >>= 1)
                ps += __shfl_xor(ps, off, 32);
            lrow[r] = lrow[r] * alpha + ps;
            mrow[r] = mnew;
#pragma unroll
            for (int nc = 0; nc < 4; ++nc) acc[nc][r] *= alpha;

            // stage P (C layout) into LDS as f16
            pl[(r + 8 * half) * 32 + lr]      = (h16)p0;
            pl[(r + 8 * half) * 32 + 16 + lr] = (h16)p1;
        }

        // ---- reload P as an A fragment (16x32) from LDS ----
        v16h pa;
        {
            const h16* pp = pl + lr * 32 + half * 8;
            v8h lo = *(const v8h*)pp;
            v8h hi = *(const v8h*)(pp + 16);
#pragma unroll
            for (int i = 0; i < 8; ++i) { pa[i] = lo[i]; pa[8 + i] = hi[i]; }
        }

        // ---- ctx += P(16x32) @ V(32x64) : four 16-wide N tiles ----
#pragma unroll
        for (int nc = 0; nc < 4; ++nc) {
            v16h vb = load_b_frag(Vp, HDIM, j0, nc * 16, lane);
            acc[nc] = wmma16x16x32(pa, vb, acc[nc]);
        }
    }

    // ---- normalize and store ctx in [B*S][1024] f16 (row-major) ----
#pragma unroll
    for (int nc = 0; nc < 4; ++nc) {
        const int col = h * HDIM + nc * 16 + lr;
#pragma unroll
        for (int r = 0; r < 8; ++r) {
            const int row = b * SEQ + q0 + r + 8 * half;
            ctxb[(size_t)row * D_OUT + col] = (h16)(acc[nc][r] / lrow[r]);
        }
    }
}

// ---------------------------- output projection ----------------------------
// ctx [4096][1024] f16 @ wprojT [1024][1024] f16 + bias -> f32 d_out
// Same 32x64 register blocking as the QKV GEMM.

__global__ __launch_bounds__(128) void proj_gemm_kernel(
        const h16* __restrict__ ctxb, const h16* __restrict__ wT,
        const float* __restrict__ bias, float* __restrict__ out) {
    const int wave = threadIdx.x >> 5;
    const int lane = threadIdx.x & 31;
    const int n0 = blockIdx.x * 64;
    const int m0 = blockIdx.y * 128 + wave * 32;

    v8f acc[2][4] = {};
    for (int k = 0; k < D_OUT; k += 32) {
        v16h a0 = load_a_frag(ctxb, D_OUT, m0,      k, lane);
        v16h a1 = load_a_frag(ctxb, D_OUT, m0 + 16, k, lane);
#pragma unroll
        for (int j = 0; j < 4; ++j) {
            v16h b = load_b_frag(wT, D_OUT, k, n0 + 16 * j, lane);
            acc[0][j] = wmma16x16x32(a0, b, acc[0][j]);
            acc[1][j] = wmma16x16x32(a1, b, acc[1][j]);
        }
    }

    const int half = lane >> 4, lr = lane & 15;
#pragma unroll
    for (int i = 0; i < 2; ++i) {
        const int mb = m0 + i * 16;
#pragma unroll
        for (int j = 0; j < 4; ++j) {
            const int n = n0 + 16 * j + lr;
            const float bn = bias[n];
#pragma unroll
            for (int r = 0; r < 8; ++r) {
                const int m = mb + r + 8 * half;
                out[(size_t)m * D_OUT + n] = acc[i][j][r] + bn;
            }
        }
    }
}

// ---------------------------------------------------------------------------

extern "C" void kernel_launch(void* const* d_in, const int* in_sizes, int n_in,
                              void* d_out, int out_size, void* d_ws, size_t ws_size,
                              hipStream_t stream) {
    const float* x      = (const float*)d_in[0];   // [2,2048,1024]
    const float* w_qkv  = (const float*)d_in[1];   // [3072,1024]
    const float* w_proj = (const float*)d_in[2];   // [1024,1024]
    const float* b_proj = (const float*)d_in[3];   // [1024]
    float* out = (float*)d_out;                    // [2,2048,1024] f32

    char* ws = (char*)d_ws;
    const size_t MB = (size_t)1 << 20;
    h16* xb     = (h16*)(ws);              //  8 MiB  x as f16 [4096][1024]
    h16* wqkvT  = (h16*)(ws +  8 * MB);    //  6 MiB  [1024][3072]
    h16* wprojT = (h16*)(ws + 14 * MB);    //  2 MiB  [1024][1024]
    h16* Qb     = (h16*)(ws + 16 * MB);    //  8 MiB  [B,H,S,64]
    h16* Ktb    = (h16*)(ws + 24 * MB);    //  8 MiB  [B,H,64,S]
    h16* Vb     = (h16*)(ws + 32 * MB);    //  8 MiB  [B,H,S,64]
    h16* ctxb   = (h16*)(ws + 40 * MB);    //  8 MiB  [4096][1024]
    // total workspace: 48 MiB

    convert_f32_to_f16_kernel<<<(MROWS * D_IN + 255) / 256, 256, 0, stream>>>(
        x, xb, MROWS * D_IN);
    transpose_to_f16_kernel<NQKV, D_IN><<<(NQKV * D_IN + 255) / 256, 256, 0, stream>>>(
        w_qkv, wqkvT);
    transpose_to_f16_kernel<D_OUT, D_OUT><<<(D_OUT * D_OUT + 255) / 256, 256, 0, stream>>>(
        w_proj, wprojT);

    qkv_gemm_kernel<<<dim3(NQKV / 64, MROWS / 128), 128, 0, stream>>>(
        xb, wqkvT, Qb, Ktb, Vb);

    attn_kernel<<<(BATCH * NHEADS * (SEQ / 16)) / 4, 128, 0, stream>>>(
        Qb, Ktb, Vb, ctxb);

    proj_gemm_kernel<<<dim3(D_OUT / 64, MROWS / 128), 128, 0, stream>>>(
        ctxb, wprojT, b_proj, out);
}
